// Decoder_77403900608529
// MI455X (gfx1250) — matrix operand order repaired and checked by reading, not compiled, not run on hardware
//
#include <hip/hip_runtime.h>
#include <hip/hip_bf16.h>

// ---------------------------------------------------------------------------
// Recurrent decoder, MI455X (gfx1250, wave32, WMMA).
// B=32768 rows, T=30 steps, D=384, FF=1024, HID=64.
// Compute ~2.2 TFLOP, HBM ~100MB -> WMMA bf16 w/ f32 accumulate.
// Weight traffic from L2 is the binding resource: each wave therefore owns
// BOTH 16-row strips of its workgroup tile (M=32 per wave) and a narrow
// column group, so every B-fragment fetched from L2 feeds 2 WMMAs
// (halves L2 weight traffic vs 1 strip/wave).
// Anti-LICM: a laundered zero offset per t-iteration keeps kernarg
// provenance (global_load_b128, not flat) while preventing cross-step
// hoisting of weight loads (which previously spilled to scratch).
// ---------------------------------------------------------------------------

#define B_SZ   32768
#define T_SZ   30
#define D_SZ   384
#define FF_SZ  1024
#define HID_SZ 64
#define ROWS   32          // batch rows per workgroup
#define NTHR   256         // 8 wave32s
#define LN_EPS 1e-5f

typedef __attribute__((ext_vector_type(16))) __bf16        v16bf;
typedef __attribute__((ext_vector_type(8)))  float         v8f;
typedef __attribute__((ext_vector_type(4)))  unsigned int  u32x4;

__device__ __forceinline__ unsigned short f2bf(float f) {
  unsigned int u = __builtin_bit_cast(unsigned int, f);
  u += 0x7fffu + ((u >> 16) & 1u);              // round-to-nearest-even
  return (unsigned short)(u >> 16);
}

// Returns 0 in an SGPR, opaque to the optimizer.
__device__ __forceinline__ int anti_licm_zero() {
  int z = 0;
  asm volatile("" : "+s"(z));
  return z;
}

// Load a 16-element bf16 fragment as two 16B vectors: q0 at p, q1 at p+off.
// A-frag (16x32, per CDNA5 ISA): lane(0-15)=row M, dwords0-3 K=half*8+0..7,
// dwords4-7 K=16+half*8+0..7  -> off = 16 ushorts.
// B-frag (32x16): lane holds col N, same K layout as A-frag
// -> off = 16 ushorts.
__device__ __forceinline__ v16bf frag_ld2(const unsigned short* p, int off) {
  union { v16bf v; u32x4 q[2]; } f;
  f.q[0] = *(const u32x4*)(p);
  f.q[1] = *(const u32x4*)(p + off);
  return f.v;
}

__device__ __forceinline__ float wave_sum(float v) {
  #pragma unroll
  for (int off = 16; off > 0; off >>= 1) v += __shfl_xor(v, off, 32);
  return v;
}

// --------------------------- weight prep (fp32 -> bf16) --------------------
__global__ void cvt_f32_bf16(const float* __restrict__ src,
                             unsigned short* __restrict__ dst, int n) {
  int i = blockIdx.x * blockDim.x + threadIdx.x;
  int stride = gridDim.x * blockDim.x;
  for (; i < n; i += stride) dst[i] = f2bf(src[i]);
}

// ------------------------------- main kernel -------------------------------
__global__ __launch_bounds__(NTHR)
void decoder_gfx1250_kernel(
    const float* __restrict__ init_hidden, const float* __restrict__ plan,
    const float* __restrict__ gate,        const float* __restrict__ init_state,
    const float* __restrict__ Wp,          const float* __restrict__ bp,
    const float* __restrict__ Ws,          const float* __restrict__ bs,
    const float* __restrict__ bqkv,        const float* __restrict__ bo,
    const float* __restrict__ g1,          const float* __restrict__ beta1,
    const float* __restrict__ g2,          const float* __restrict__ beta2,
    const float* __restrict__ b1,          const float* __restrict__ b2,
    const float* __restrict__ bd1,         const float* __restrict__ Wd2,
    const float* __restrict__ bd2,
    const unsigned short* __restrict__ wv_bf,
    const unsigned short* __restrict__ wo_bf,
    const unsigned short* __restrict__ w1_bf,
    const unsigned short* __restrict__ w2_bf,
    const unsigned short* __restrict__ wd1_bf,
    float* __restrict__ out) {

  // LDS: ~145 KB of the 320 KB WGP budget (2 workgroups / WGP).
  __shared__ __align__(16) float          yf[ROWS * D_SZ];   // fp32 activation (residual/LN)
  __shared__ __align__(16) unsigned short xb[ROWS * D_SZ];   // bf16 activation (WMMA A)
  __shared__ __align__(16) unsigned short hb[ROWS * FF_SZ];  // bf16 FF intermediate / v-buffer
  __shared__ __align__(16) float          uf[ROWS * HID_SZ]; // head intermediate fp32
  __shared__ float st[ROWS * 3];
  __shared__ float pl[ROWS * 3];
  __shared__ float gt[ROWS];

  const int tid  = threadIdx.x;
  const int lane = tid & 31;
  const int wave = tid >> 5;       // == column group for the GEMMs
  const int hlf  = lane >> 4;      // half-wave select (fragment layouts)
  const int l16  = lane & 15;
  const int row0 = blockIdx.x * ROWS;

  const v8f vzero = {0.f,0.f,0.f,0.f,0.f,0.f,0.f,0.f};

  if (tid < ROWS * 3) st[tid] = init_state[(size_t)row0 * 3 + tid];
  if (tid < ROWS)     gt[tid] = gate[row0 + tid];
  // Warm L2 for the bf16 weight streams (global_prefetch_b8 path).
  __builtin_prefetch(wv_bf + (size_t)tid * 64, 0, 3);
  __builtin_prefetch(w1_bf + (size_t)tid * 64, 0, 3);
  __builtin_prefetch(w2_bf + (size_t)tid * 64, 0, 3);
  __syncthreads();

  #pragma unroll 1
  for (int t = 0; t < T_SZ; ++t) {
    // Opaque zero, new SSA value each iteration: blocks cross-t hoisting.
    const int zo = anti_licm_zero();
    const float* ih_t   = init_hidden + zo;
    const float* pln_t  = plan + zo;
    const float* Wp_t   = Wp + zo;
    const float* bp_t   = bp + zo;
    const float* Ws_t   = Ws + zo;
    const float* bs_t   = bs + zo;
    const float* bv_t   = bqkv + 2 * D_SZ + zo;
    const float* bo_t   = bo + zo;
    const float* g1_t   = g1 + zo;
    const float* be1_t  = beta1 + zo;
    const float* g2_t   = g2 + zo;
    const float* be2_t  = beta2 + zo;
    const float* b1_t   = b1 + zo;
    const float* b2_t   = b2 + zo;
    const float* bd1_t  = bd1 + zo;
    const float* Wd2_t  = Wd2 + zo;
    const float* bd2_t  = bd2 + zo;
    const unsigned short* wv_t  = wv_bf + zo;
    const unsigned short* wo_t  = wo_bf + zo;
    const unsigned short* w1_t  = w1_bf + zo;
    const unsigned short* w2_t  = w2_bf + zo;
    const unsigned short* wd1_t = wd1_bf + zo;

    // ---- stage plan_t for this tile ------------------------------------
    if (tid < ROWS * 3) {
      int m = tid / 3, c = tid - m * 3;
      pl[tid] = pln_t[(size_t)(row0 + m) * (T_SZ * 3) + t * 3 + c];
    }
    __syncthreads();

    // ---- phase 1: x0 = state@Ws.T + bs + (plan@Wp.T + bp)*gate + h0 ----
    for (int idx = tid; idx < ROWS * D_SZ; idx += NTHR) {
      int m = idx / D_SZ, d = idx - m * D_SZ;
      float pf = pl[m*3]*Wp_t[d*3] + pl[m*3+1]*Wp_t[d*3+1] + pl[m*3+2]*Wp_t[d*3+2] + bp_t[d];
      float sf = st[m*3]*Ws_t[d*3] + st[m*3+1]*Ws_t[d*3+1] + st[m*3+2]*Ws_t[d*3+2] + bs_t[d];
      float x  = sf + pf * gt[m] + ih_t[(size_t)(row0 + m) * D_SZ + d];
      yf[idx] = x;
      xb[idx] = f2bf(x);
    }
    __syncthreads();

    // ---- GEMM1: v = x0 @ Wv.T + bv   (M=32,N=384,K=384) ----------------
    // Wave owns 3 N-tiles x both row strips: B reused for 2 WMMAs.
    {
      v8f acc[2][3];
      #pragma unroll
      for (int s = 0; s < 2; ++s)
        #pragma unroll
        for (int j = 0; j < 3; ++j) acc[s][j] = vzero;
      for (int kt = 0; kt < 12; ++kt) {
        v16bf a0 = frag_ld2(&xb[(l16)      * D_SZ + kt*32 + hlf*8], 16);
        v16bf a1 = frag_ld2(&xb[(16 + l16) * D_SZ + kt*32 + hlf*8], 16);
        v16bf bfr[3];
        #pragma unroll
        for (int j = 0; j < 3; ++j)
          bfr[j] = frag_ld2(&wv_t[(size_t)(wave*48 + j*16 + l16) * D_SZ + kt*32 + hlf*8], 16);
        #pragma unroll
        for (int j = 0; j < 3; ++j) {
          acc[0][j] = __builtin_amdgcn_wmma_f32_16x16x32_bf16(
                          false, a0, false, bfr[j], (short)0, acc[0][j], false, false);
          acc[1][j] = __builtin_amdgcn_wmma_f32_16x16x32_bf16(
                          false, a1, false, bfr[j], (short)0, acc[1][j], false, false);
        }
      }
      #pragma unroll
      for (int j = 0; j < 3; ++j) {
        int n = wave*48 + j*16 + l16;
        float bn = bv_t[n];
        #pragma unroll
        for (int s = 0; s < 2; ++s)
          #pragma unroll
          for (int r = 0; r < 8; ++r) {
            int m = s*16 + hlf*8 + r;
            hb[m * D_SZ + n] = f2bf(acc[s][j][r] + bn);   // v-buffer (stride D)
          }
      }
    }
    __syncthreads();

    // ---- GEMM2: attn = v @ Wo.T + bo ; yf += attn ----------------------
    {
      v8f acc[2][3];
      #pragma unroll
      for (int s = 0; s < 2; ++s)
        #pragma unroll
        for (int j = 0; j < 3; ++j) acc[s][j] = vzero;
      for (int kt = 0; kt < 12; ++kt) {
        v16bf a0 = frag_ld2(&hb[(l16)      * D_SZ + kt*32 + hlf*8], 16);
        v16bf a1 = frag_ld2(&hb[(16 + l16) * D_SZ + kt*32 + hlf*8], 16);
        v16bf bfr[3];
        #pragma unroll
        for (int j = 0; j < 3; ++j)
          bfr[j] = frag_ld2(&wo_t[(size_t)(wave*48 + j*16 + l16) * D_SZ + kt*32 + hlf*8], 16);
        #pragma unroll
        for (int j = 0; j < 3; ++j) {
          acc[0][j] = __builtin_amdgcn_wmma_f32_16x16x32_bf16(
                          false, a0, false, bfr[j], (short)0, acc[0][j], false, false);
          acc[1][j] = __builtin_amdgcn_wmma_f32_16x16x32_bf16(
                          false, a1, false, bfr[j], (short)0, acc[1][j], false, false);
        }
      }
      #pragma unroll
      for (int j = 0; j < 3; ++j) {
        int n = wave*48 + j*16 + l16;
        float bn = bo_t[n];
        #pragma unroll
        for (int s = 0; s < 2; ++s)
          #pragma unroll
          for (int r = 0; r < 8; ++r) {
            int m = s*16 + hlf*8 + r;
            yf[m * D_SZ + n] += acc[s][j][r] + bn;
          }
      }
    }
    __syncthreads();

    // ---- LN1 -> xb (bf16 WMMA operand) and yf (fp32 residual) ----------
    #pragma unroll
    for (int i = 0; i < ROWS / 8; ++i) {
      int m = wave * (ROWS / 8) + i;
      float s = 0.f;
      for (int d = lane; d < D_SZ; d += 32) s += yf[m * D_SZ + d];
      float mean = wave_sum(s) * (1.0f / D_SZ);
      float v = 0.f;
      for (int d = lane; d < D_SZ; d += 32) {
        float df = yf[m * D_SZ + d] - mean; v += df * df;
      }
      float inv = rsqrtf(wave_sum(v) * (1.0f / D_SZ) + LN_EPS);
      for (int d = lane; d < D_SZ; d += 32) {
        float xn = (yf[m * D_SZ + d] - mean) * inv * g1_t[d] + be1_t[d];
        yf[m * D_SZ + d] = xn;
        xb[m * D_SZ + d] = f2bf(xn);
      }
    }
    __syncthreads();

    // ---- FF1: h = relu(x1 @ W1.T + b1) (N=1024; 8 tiles/wave, 2 chunks) -
    {
      #pragma unroll 1
      for (int c = 0; c < 2; ++c) {
        v8f acc[2][4];
        #pragma unroll
        for (int s = 0; s < 2; ++s)
          #pragma unroll
          for (int j = 0; j < 4; ++j) acc[s][j] = vzero;
        for (int kt = 0; kt < 12; ++kt) {
          v16bf a0 = frag_ld2(&xb[(l16)      * D_SZ + kt*32 + hlf*8], 16);
          v16bf a1 = frag_ld2(&xb[(16 + l16) * D_SZ + kt*32 + hlf*8], 16);
          v16bf bfr[4];
          #pragma unroll
          for (int j = 0; j < 4; ++j)
            bfr[j] = frag_ld2(&w1_t[(size_t)(wave*128 + c*64 + j*16 + l16) * D_SZ + kt*32 + hlf*8], 16);
          #pragma unroll
          for (int j = 0; j < 4; ++j) {
            acc[0][j] = __builtin_amdgcn_wmma_f32_16x16x32_bf16(
                            false, a0, false, bfr[j], (short)0, acc[0][j], false, false);
            acc[1][j] = __builtin_amdgcn_wmma_f32_16x16x32_bf16(
                            false, a1, false, bfr[j], (short)0, acc[1][j], false, false);
          }
        }
        #pragma unroll
        for (int j = 0; j < 4; ++j) {
          int n = wave*128 + c*64 + j*16 + l16;
          float bn = b1_t[n];
          #pragma unroll
          for (int s = 0; s < 2; ++s)
            #pragma unroll
            for (int r = 0; r < 8; ++r) {
              int m = s*16 + hlf*8 + r;
              hb[m * FF_SZ + n] = f2bf(fmaxf(acc[s][j][r] + bn, 0.f));
            }
        }
      }
    }
    __syncthreads();

    // ---- FF2: yf += h @ W2.T + b2   (K=1024) ---------------------------
    {
      v8f acc[2][3];
      #pragma unroll
      for (int s = 0; s < 2; ++s)
        #pragma unroll
        for (int j = 0; j < 3; ++j) acc[s][j] = vzero;
      for (int kt = 0; kt < 32; ++kt) {
        v16bf a0 = frag_ld2(&hb[(l16)      * FF_SZ + kt*32 + hlf*8], 16);
        v16bf a1 = frag_ld2(&hb[(16 + l16) * FF_SZ + kt*32 + hlf*8], 16);
        v16bf bfr[3];
        #pragma unroll
        for (int j = 0; j < 3; ++j)
          bfr[j] = frag_ld2(&w2_t[(size_t)(wave*48 + j*16 + l16) * FF_SZ + kt*32 + hlf*8], 16);
        #pragma unroll
        for (int j = 0; j < 3; ++j) {
          acc[0][j] = __builtin_amdgcn_wmma_f32_16x16x32_bf16(
                          false, a0, false, bfr[j], (short)0, acc[0][j], false, false);
          acc[1][j] = __builtin_amdgcn_wmma_f32_16x16x32_bf16(
                          false, a1, false, bfr[j], (short)0, acc[1][j], false, false);
        }
      }
      #pragma unroll
      for (int j = 0; j < 3; ++j) {
        int n = wave*48 + j*16 + l16;
        float bn = b2_t[n];
        #pragma unroll
        for (int s = 0; s < 2; ++s)
          #pragma unroll
          for (int r = 0; r < 8; ++r) {
            int m = s*16 + hlf*8 + r;
            yf[m * D_SZ + n] += acc[s][j][r] + bn;
          }
      }
    }
    __syncthreads();

    // ---- LN2 -> xb ------------------------------------------------------
    #pragma unroll
    for (int i = 0; i < ROWS / 8; ++i) {
      int m = wave * (ROWS / 8) + i;
      float s = 0.f;
      for (int d = lane; d < D_SZ; d += 32) s += yf[m * D_SZ + d];
      float mean = wave_sum(s) * (1.0f / D_SZ);
      float v = 0.f;
      for (int d = lane; d < D_SZ; d += 32) {
        float df = yf[m * D_SZ + d] - mean; v += df * df;
      }
      float inv = rsqrtf(wave_sum(v) * (1.0f / D_SZ) + LN_EPS);
      for (int d = lane; d < D_SZ; d += 32) {
        float xn = (yf[m * D_SZ + d] - mean) * inv * g2_t[d] + be2_t[d];
        xb[m * D_SZ + d] = f2bf(xn);
      }
    }
    __syncthreads();

    // ---- head GEMM: u = elu(x2 @ Wd1.T + bd1) (N=64; waves 0-3) --------
    if (wave < 4) {
      v8f acc0 = vzero, acc1 = vzero;
      for (int kt = 0; kt < 12; ++kt) {
        v16bf a0 = frag_ld2(&xb[(l16)      * D_SZ + kt*32 + hlf*8], 16);
        v16bf a1 = frag_ld2(&xb[(16 + l16) * D_SZ + kt*32 + hlf*8], 16);
        v16bf b  = frag_ld2(&wd1_t[(size_t)(wave*16 + l16) * D_SZ + kt*32 + hlf*8], 16);
        acc0 = __builtin_amdgcn_wmma_f32_16x16x32_bf16(
                   false, a0, false, b, (short)0, acc0, false, false);
        acc1 = __builtin_amdgcn_wmma_f32_16x16x32_bf16(
                   false, a1, false, b, (short)0, acc1, false, false);
      }
      int n = wave*16 + l16;
      float bn = bd1_t[n];
      #pragma unroll
      for (int r = 0; r < 8; ++r) {
        int m0 = hlf*8 + r;
        float e0 = acc0[r] + bn;
        uf[m0 * HID_SZ + n] = (e0 > 0.f) ? e0 : (expf(e0) - 1.f);
        float e1 = acc1[r] + bn;
        uf[(16 + m0) * HID_SZ + n] = (e1 > 0.f) ? e1 : (expf(e1) - 1.f);
      }
    }
    __syncthreads();

    // ---- upd = u @ Wd2.T + bd2 ; state += upd ; emit output ------------
    if (tid < ROWS * 3) {
      int m = tid / 3, c = tid - m * 3;
      float s = bd2_t[c];
      for (int k = 0; k < HID_SZ; ++k)
        s += uf[m * HID_SZ + k] * Wd2_t[c * HID_SZ + k];
      float ns = st[tid] + s;
      st[tid] = ns;
      out[(size_t)(row0 + m) * (T_SZ * 3) + t * 3 + c] = ns;
    }
    __syncthreads();
  }
}

// ------------------------------ launch -------------------------------------
extern "C" void kernel_launch(void* const* d_in, const int* in_sizes, int n_in,
                              void* d_out, int out_size, void* d_ws, size_t ws_size,
                              hipStream_t stream) {
  (void)in_sizes; (void)n_in; (void)out_size; (void)ws_size;
  const float* init_hidden = (const float*)d_in[0];
  const float* plan        = (const float*)d_in[1];
  const float* gate        = (const float*)d_in[2];
  const float* init_state  = (const float*)d_in[3];
  const float* Wp          = (const float*)d_in[4];
  const float* bp          = (const float*)d_in[5];
  const float* Ws          = (const float*)d_in[6];
  const float* bs          = (const float*)d_in[7];
  const float* Wqkv        = (const float*)d_in[8];
  const float* bqkv        = (const float*)d_in[9];
  const float* Wo          = (const float*)d_in[10];
  const float* bo          = (const float*)d_in[11];
  const float* g1          = (const float*)d_in[12];
  const float* beta1       = (const float*)d_in[13];
  const float* g2          = (const float*)d_in[14];
  const float* beta2       = (const float*)d_in[15];
  const float* W1          = (const float*)d_in[16];
  const float* b1          = (const float*)d_in[17];
  const float* W2          = (const float*)d_in[18];
  const float* b2          = (const float*)d_in[19];
  const float* Wd1         = (const float*)d_in[20];
  const float* bd1         = (const float*)d_in[21];
  const float* Wd2         = (const float*)d_in[22];
  const float* bd2         = (const float*)d_in[23];
  float* out = (float*)d_out;

  // bf16 weight cache in workspace
  unsigned short* ws  = (unsigned short*)d_ws;
  unsigned short* wv  = ws;                                   // 384*384
  unsigned short* wo  = ws + D_SZ * D_SZ;                     // 384*384
  unsigned short* w1  = wo + D_SZ * D_SZ;                     // 1024*384
  unsigned short* w2  = w1 + FF_SZ * D_SZ;                    // 384*1024
  unsigned short* wd1 = w2 + D_SZ * FF_SZ;                    // 64*384

  cvt_f32_bf16<<<144, 256, 0, stream>>>(Wqkv + 2 * D_SZ * D_SZ, wv, D_SZ * D_SZ);
  cvt_f32_bf16<<<144, 256, 0, stream>>>(Wo,  wo,  D_SZ * D_SZ);
  cvt_f32_bf16<<<384, 256, 0, stream>>>(W1,  w1,  FF_SZ * D_SZ);
  cvt_f32_bf16<<<384, 256, 0, stream>>>(W2,  w2,  D_SZ * FF_SZ);
  cvt_f32_bf16<<<24,  256, 0, stream>>>(Wd1, wd1, HID_SZ * D_SZ);

  decoder_gfx1250_kernel<<<B_SZ / ROWS, NTHR, 0, stream>>>(
      init_hidden, plan, gate, init_state, Wp, bp, Ws, bs, bqkv, bo,
      g1, beta1, g2, beta2, b1, b2, bd1, Wd2, bd2,
      wv, wo, w1, w2, wd1, out);
}
